// FEMBATinyFullInput_79182017069422
// MI455X (gfx1250) — matrix operand order repaired and checked by reading, not compiled, not run on hardware
//
#include <hip/hip_runtime.h>
#include <hip/hip_bf16.h>
#include <math.h>

// ---------------- model constants ----------------
#define Bn    16
#define Hh    22
#define Wi    20480
#define EMB   35
#define GHn   11
#define Lseq  1280
#define DMod  385           // GHn * EMB
#define DIn   256
#define Nst   4
#define DTR   25
#define Mtok  (Bn * Lseq)   // 20480

typedef __attribute__((ext_vector_type(16))) _Float16 v16h;
typedef __attribute__((ext_vector_type(8)))  float    v8f;

static inline int cdiv(int a, int b) { return (a + b - 1) / b; }

// ---------------- device helpers ----------------
__device__ inline float slot_scale(const float* slot) {
    return fmaxf(*slot, 1e-8f) / 127.f;
}
__device__ inline float dq(float v, float s) {
    // fake-quant forward: clip(round(v/s), -128, 127) * s (round-to-nearest-even)
    return fminf(fmaxf(rintf(v / s), -128.f), 127.f) * s;
}
__device__ inline void wave_amax_atomic(float m, float* slot) {
#pragma unroll
    for (int off = 16; off; off >>= 1)
        m = fmaxf(m, __shfl_xor(m, off, 32));
    if ((threadIdx.x & 31) == 0)
        atomicMax((unsigned int*)slot, __float_as_uint(m));
}
__device__ inline float silu_f(float v) { return v / (1.f + expf(-v)); }
__device__ inline unsigned pack2h(float a, float b) {
    unsigned short ha = __builtin_bit_cast(unsigned short, (_Float16)a);
    unsigned short hb = __builtin_bit_cast(unsigned short, (_Float16)b);
    return (unsigned)ha | ((unsigned)hb << 16);
}

// ---------------- small utility kernels ----------------
__global__ void zero_slots_kernel(float* slots) {
    if (threadIdx.x < 16) slots[threadIdx.x] = 0.f;
}

__global__ void amax_kernel(const float* __restrict__ p, size_t n, float* slot) {
    size_t i = (size_t)blockIdx.x * blockDim.x + threadIdx.x;
    size_t stride = (size_t)gridDim.x * blockDim.x;
    float m = 0.f;
    for (; i < n; i += stride) m = fmaxf(m, fabsf(p[i]));
    wave_amax_atomic(m, slot);
}

__global__ void quant_kernel(float* __restrict__ p, size_t n, const float* slot) {
    const float s = slot_scale(slot);
    size_t i = (size_t)blockIdx.x * blockDim.x + threadIdx.x;
    size_t stride = (size_t)gridDim.x * blockDim.x;
    for (; i < n; i += stride) p[i] = dq(p[i], s);
}

// ---------------- patch embed (fused input-quant + weight-quant + conv2d) ----------------
__global__ __launch_bounds__(256)
void patch_embed_kernel(const float* __restrict__ x, const float* __restrict__ pw,
                        const float* __restrict__ pb, const float* slot_x,
                        const float* slot_w, float* __restrict__ X, float* slot_out) {
    const float sx = slot_scale(slot_x);
    const float sw = slot_scale(slot_w);
    long idx = (long)blockIdx.x * 256 + threadIdx.x;
    float m = 0.f;
    if (idx < (long)Mtok * DMod) {
        int dm = (int)(idx % DMod);
        long tmp = idx / DMod;
        int t = (int)(tmp % Lseq);
        int b = (int)(tmp / Lseq);
        int gh = dm / EMB, e = dm % EMB;
        float acc = pb[e];
#pragma unroll
        for (int ph = 0; ph < 2; ++ph)
#pragma unroll
            for (int pwi = 0; pwi < 16; ++pwi) {
                float xv = dq(x[((size_t)b * Hh + (gh * 2 + ph)) * Wi + (size_t)t * 16 + pwi], sx);
                float wv = dq(pw[(e * 2 + ph) * 16 + pwi], sw);
                acc += xv * wv;
            }
        X[idx] = acc;
        m = fabsf(acc);
    }
    wave_amax_atomic(m, slot_out);
}

// X[i] += fq(pos)[i % (L*DM)] ; accumulate amax of result
__global__ __launch_bounds__(256)
void add_pos_kernel(float* __restrict__ X, const float* __restrict__ pos,
                    const float* slot_pos, float* slot_out) {
    const float sp = slot_scale(slot_pos);
    long idx = (long)blockIdx.x * 256 + threadIdx.x;
    float m = 0.f;
    if (idx < (long)Mtok * DMod) {
        float v = X[idx] + dq(pos[idx % ((long)Lseq * DMod)], sp);
        X[idx] = v;
        m = fabsf(v);
    }
    wave_amax_atomic(m, slot_out);
}

// ---------------- generic WMMA GEMM: C[M,N] = A[M,K] (lda) x B[K,N] ----------------
// Workgroup tile 64(M) x 64(N), 8 waves; each wave: one A frag, two B frags,
// two v_wmma_f32_16x16x32_f16 per K-step. Assumes M % 64 == 0 (true for all calls).
// MODE: 0 = store ; 2 = softplus(+bias) ; 3 = flip-row accumulate.
#define TM 64
#define TN 64
#define TK 32
#define LDSW 20            // row stride in uints (80B, 16B aligned)

typedef union { uint4 u4[2]; v16h h16; } frag_u;

template <bool TAIL>
__device__ __forceinline__ void stage_and_mma(
    const float* __restrict__ A, int lda, const float* __restrict__ Bw,
    int Kd, int N, int k0, int blockM, int blockN,
    unsigned (&As32)[TM][LDSW], unsigned (&Bs32)[TN][LDSW],
    v8f& acc0, v8f& acc1) {
    const int tid   = threadIdx.x;
    const int lane  = tid & 31;
    const int wave  = tid >> 5;
    const int mtile = wave & 3;       // 4 M-subtiles of 16
    const int nhalf = wave >> 2;      // 2 N-subtiles of 32
    const int half  = lane >> 4;
    const int l15   = lane & 15;

    // ---- stage A tile 64x32 (1024 f16-pairs, 4 per thread), branchless ----
    float fa[8];
#pragma unroll
    for (int i = 0; i < 4; ++i) {
        int p  = tid + i * 256;          // pair index; consecutive tid -> consec k
        int r  = p >> 4;                 // m row
        int c  = (p & 15) * 2;           // k within tile
        int gm = blockM + r;             // M % 64 == 0 -> in range
        int g0 = k0 + c;
        size_t base = (size_t)gm * lda;
        if (TAIL) {
            fa[2 * i + 0] = A[base + min(g0,     Kd - 1)];
            fa[2 * i + 1] = A[base + min(g0 + 1, Kd - 1)];
            fa[2 * i + 0] = (g0     < Kd) ? fa[2 * i + 0] : 0.f;
            fa[2 * i + 1] = (g0 + 1 < Kd) ? fa[2 * i + 1] : 0.f;
        } else {
            fa[2 * i + 0] = A[base + g0];
            fa[2 * i + 1] = A[base + g0 + 1];
        }
    }
    // ---- stage B tile 32x64 transposed (1024 pairs, 4 per thread) ----
    float fb[8];
#pragma unroll
    for (int i = 0; i < 4; ++i) {
        int p  = tid + i * 256;
        int n  = p & 63;                 // consecutive tid -> consec n (coalesced)
        int kp = p >> 6;                 // 0..15
        int gn = min(blockN + n, N - 1);
        int g0 = k0 + kp * 2;
        int okn = (blockN + n) < N;
        if (TAIL) {
            fb[2 * i + 0] = Bw[(size_t)min(g0,     Kd - 1) * N + gn];
            fb[2 * i + 1] = Bw[(size_t)min(g0 + 1, Kd - 1) * N + gn];
            fb[2 * i + 0] = (okn && g0     < Kd) ? fb[2 * i + 0] : 0.f;
            fb[2 * i + 1] = (okn && g0 + 1 < Kd) ? fb[2 * i + 1] : 0.f;
        } else {
            fb[2 * i + 0] = Bw[(size_t)g0 * N + gn];
            fb[2 * i + 1] = Bw[(size_t)(g0 + 1) * N + gn];
            fb[2 * i + 0] = okn ? fb[2 * i + 0] : 0.f;
            fb[2 * i + 1] = okn ? fb[2 * i + 1] : 0.f;
        }
    }
#pragma unroll
    for (int i = 0; i < 4; ++i) {
        int p = tid + i * 256;
        As32[p >> 4][p & 15] = pack2h(fa[2 * i], fa[2 * i + 1]);
    }
#pragma unroll
    for (int i = 0; i < 4; ++i) {
        int p = tid + i * 256;
        Bs32[p & 63][p >> 6] = pack2h(fb[2 * i], fb[2 * i + 1]);
    }
    __syncthreads();

    // ---- fragments: pure ds_load_b128 per ISA lane layouts ----
    // A: lane(half,l15): k = (e&7) + (e>>3)*16 + half*8  -> uint runs [h*4..+3],[8+h*4..+3]
    frag_u fa_u;
    const uint4* Arow = (const uint4*)&As32[mtile * 16 + l15][0];
    fa_u.u4[0] = Arow[half];
    fa_u.u4[1] = Arow[2 + half];
    // B: lane(half,l15): k = (e&7) + (e>>3)*16 + half*8  -> uint runs [h*4..+3],[8+h*4..+3]
    frag_u fb0_u, fb1_u;
    const uint4* Brow0 = (const uint4*)&Bs32[nhalf * 32 + l15][0];
    const uint4* Brow1 = (const uint4*)&Bs32[nhalf * 32 + 16 + l15][0];
    fb0_u.u4[0] = Brow0[half];
    fb0_u.u4[1] = Brow0[2 + half];
    fb1_u.u4[0] = Brow1[half];
    fb1_u.u4[1] = Brow1[2 + half];

    acc0 = __builtin_amdgcn_wmma_f32_16x16x32_f16(false, fa_u.h16, false, fb0_u.h16,
                                                  (short)0, acc0, false, false);
    acc1 = __builtin_amdgcn_wmma_f32_16x16x32_f16(false, fa_u.h16, false, fb1_u.h16,
                                                  (short)0, acc1, false, false);
    __syncthreads();
}

template <int MODE>
__device__ __forceinline__ void gemm_store(float* __restrict__ C, int gm, int gn, int N,
                                           float r, const float* __restrict__ bias,
                                           int Ltime) {
    if (gn < N) {                       // gm always in range (M % 64 == 0)
        if (MODE == 2) {
            r += bias[gn];
            r = fmaxf(r, 0.f) + log1pf(expf(-fabsf(r)));   // stable branchless softplus
        }
        if (MODE == 3) {
            int bb = gm / Ltime, t = gm % Ltime;
            size_t dst = ((size_t)bb * Ltime + (Ltime - 1 - t)) * (size_t)N + gn;
            C[dst] += r;                // each element touched once: safe
        } else {
            C[(size_t)gm * N + gn] = r;
        }
    }
}

template <int MODE>
__global__ __launch_bounds__(256)
void gemm_wmma_kernel(const float* __restrict__ A, int lda,
                      const float* __restrict__ Bw, float* __restrict__ C,
                      int M, int Kd, int N, const float* __restrict__ bias,
                      int Ltime) {
    __shared__ __align__(16) unsigned As32[TM][LDSW];  // [m][k-pair]
    __shared__ __align__(16) unsigned Bs32[TN][LDSW];  // [n][k-pair] (transposed)

    const int tid   = threadIdx.x;
    const int lane  = tid & 31;
    const int wave  = tid >> 5;
    const int mtile = wave & 3;
    const int nhalf = wave >> 2;
    const int half  = lane >> 4;
    const int l15   = lane & 15;

    const int blockM = blockIdx.y * TM;
    const int blockN = blockIdx.x * TN;

    v8f acc0 = {0.f, 0.f, 0.f, 0.f, 0.f, 0.f, 0.f, 0.f};
    v8f acc1 = acc0;

    const int kmain = Kd & ~(TK - 1);
    for (int k0 = 0; k0 < kmain; k0 += TK)
        stage_and_mma<false>(A, lda, Bw, Kd, N, k0, blockM, blockN, As32, Bs32, acc0, acc1);
    if (kmain < Kd)
        stage_and_mma<true>(A, lda, Bw, Kd, N, kmain, blockM, blockN, As32, Bs32, acc0, acc1);

    // ---- epilogue: C lane layout n = l15, m = half*8 + v ----
    const int gnB = blockN + nhalf * 32 + l15;
#pragma unroll
    for (int v = 0; v < 8; ++v) {
        int gm = blockM + mtile * 16 + half * 8 + v;
        gemm_store<MODE>(C, gm, gnB,      N, acc0[v], bias, Ltime);
        gemm_store<MODE>(C, gm, gnB + 16, N, acc1[v], bias, Ltime);
    }
}

// ---------------- depthwise causal conv1d (K=4) + bias + SiLU ----------------
__global__ __launch_bounds__(256)
void conv_silu_kernel(const float* __restrict__ XZ, const float* __restrict__ cw,
                      const float* __restrict__ cb, int rev, float* __restrict__ XC) {
    long idx = (long)blockIdx.x * 256 + threadIdx.x;
    if (idx >= (long)Mtok * DIn) return;
    int d = (int)(idx % DIn);
    long tmp = idx / DIn;
    int t = (int)(tmp % Lseq);
    int b = (int)(tmp / Lseq);
    float acc = 0.f;
#pragma unroll
    for (int k = 0; k < 4; ++k) {
        int tl = t + k - 3;
        if (tl >= 0) {
            int tp = rev ? (Lseq - 1 - tl) : tl;
            acc += XZ[((size_t)b * Lseq + tp) * 512 + d] * cw[d * 4 + k];
        }
    }
    float v = acc + cb[d];
    XC[idx] = silu_f(v);
}

// ---------------- selective scan, fused with +Dp*xc and *silu(z) ----------------
__global__ __launch_bounds__(256)
void scan_kernel(float* __restrict__ XC, const float* __restrict__ DT,
                 const float* __restrict__ PROJ, const float* __restrict__ XZ,
                 const float* __restrict__ A_log, const float* __restrict__ Dp,
                 int rev) {
    int b = blockIdx.x;
    int d = threadIdx.x;
    const float A0 = -expf(A_log[d * 4 + 0]);
    const float A1 = -expf(A_log[d * 4 + 1]);
    const float A2 = -expf(A_log[d * 4 + 2]);
    const float A3 = -expf(A_log[d * 4 + 3]);
    const float Dd = Dp[d];
    float h0 = 0.f, h1 = 0.f, h2 = 0.f, h3 = 0.f;
    for (int t = 0; t < Lseq; ++t) {
        size_t tok = (size_t)b * Lseq + t;
        float xcv = XC[tok * DIn + d];
        float dtv = DT[tok * DIn + d];
        const float* pr = PROJ + tok * 33;
        float dx = dtv * xcv;
        h0 = expf(dtv * A0) * h0 + dx * pr[25];
        h1 = expf(dtv * A1) * h1 + dx * pr[26];
        h2 = expf(dtv * A2) * h2 + dx * pr[27];
        h3 = expf(dtv * A3) * h3 + dx * pr[28];
        float y = h0 * pr[29] + h1 * pr[30] + h2 * pr[31] + h3 * pr[32] + Dd * xcv;
        int tp = rev ? (Lseq - 1 - t) : t;
        float zv = XZ[((size_t)b * Lseq + tp) * 512 + 256 + d];
        y *= silu_f(zv);
        XC[tok * DIn + d] = y;
    }
}

// ---------------- residual + layernorm (+amax of result) ----------------
__global__ __launch_bounds__(128)
void ln_kernel(float* __restrict__ X, const float* __restrict__ S,
               const float* __restrict__ g, const float* __restrict__ bta,
               float* slot) {
    const int tok = blockIdx.x;
    const int tid = threadIdx.x;
    const size_t base = (size_t)tok * DMod;
    float vv[4];
    int cnt = 0;
    float sum = 0.f, sq = 0.f;
    for (int i = tid; i < DMod; i += 128) {
        float v = X[base + i] + S[base + i];
        vv[cnt++] = v;
        sum += v; sq += v * v;
    }
    __shared__ float sh1[128], sh2[128];
    sh1[tid] = sum; sh2[tid] = sq;
    __syncthreads();
    for (int s2 = 64; s2; s2 >>= 1) {
        if (tid < s2) { sh1[tid] += sh1[tid + s2]; sh2[tid] += sh2[tid + s2]; }
        __syncthreads();
    }
    float mean = sh1[0] / DMod;
    float var  = sh2[0] / DMod - mean * mean;
    float rinv = rsqrtf(var + 1e-5f);
    __syncthreads();
    float m = 0.f;
    cnt = 0;
    for (int i = tid; i < DMod; i += 128) {
        float o = (vv[cnt++] - mean) * rinv * g[i] + bta[i];
        X[base + i] = o;
        m = fmaxf(m, fabsf(o));
    }
    sh1[tid] = m;
    __syncthreads();
    for (int s2 = 64; s2; s2 >>= 1) {
        if (tid < s2) sh1[tid] = fmaxf(sh1[tid], sh1[tid + s2]);
        __syncthreads();
    }
    if (tid == 0) atomicMax((unsigned int*)slot, __float_as_uint(sh1[0]));
}

// ---------------- mean pool over L ----------------
__global__ void pool_kernel(const float* __restrict__ X, float* __restrict__ P) {
    int idx = blockIdx.x * blockDim.x + threadIdx.x;
    if (idx >= Bn * DMod) return;
    int b = idx / DMod, dm = idx % DMod;
    float s = 0.f;
    for (int t = 0; t < Lseq; ++t) s += X[((size_t)b * Lseq + t) * DMod + dm];
    P[idx] = s * (1.f / Lseq);
}

// ---------------- classifier (fused cls_w quant) ----------------
__global__ void cls_kernel(const float* __restrict__ P, const float* __restrict__ cw,
                           const float* __restrict__ cb, const float* slot_w,
                           float* __restrict__ out) {
    int tid = threadIdx.x;
    if (tid >= Bn * 2) return;
    const float s = slot_scale(slot_w);
    int b = tid >> 1, c = tid & 1;
    float acc = cb[c];
    for (int dm = 0; dm < DMod; ++dm)
        acc += P[b * DMod + dm] * dq(cw[dm * 2 + c], s);
    out[tid] = acc;
}

// ---------------- launch sequence ----------------
extern "C" void kernel_launch(void* const* d_in, const int* in_sizes, int n_in,
                              void* d_out, int out_size, void* d_ws, size_t ws_size,
                              hipStream_t stream) {
    (void)in_sizes; (void)n_in; (void)out_size; (void)ws_size;
    const float* x     = (const float*)d_in[0];
    const float* pe_w  = (const float*)d_in[1];
    const float* pe_b  = (const float*)d_in[2];
    const float* pos   = (const float*)d_in[3];
    const float* in_w  = (const float*)d_in[4];
    const float* cw    = (const float*)d_in[5];
    const float* cb    = (const float*)d_in[6];
    const float* xp_w  = (const float*)d_in[7];
    const float* dt_w  = (const float*)d_in[8];
    const float* dt_b  = (const float*)d_in[9];
    const float* A_log = (const float*)d_in[10];
    const float* Dp    = (const float*)d_in[11];
    const float* out_w = (const float*)d_in[12];
    const float* ln_g  = (const float*)d_in[13];
    const float* ln_b  = (const float*)d_in[14];
    const float* cls_w = (const float*)d_in[15];
    const float* cls_b = (const float*)d_in[16];
    float* out = (float*)d_out;

    // workspace layout (floats)
    float* ws   = (float*)d_ws;
    float* SLOT = ws;                          // 16 amax slots
    float* X    = ws + 256;                    // [M, DM]      7,884,800
    float* XZ   = X + (size_t)Mtok * DMod;     // [M, 512]    10,485,760
    float* XC   = XZ + (size_t)Mtok * 512;     // [M, 256]     5,242,880
    float* PROJ = XC + (size_t)Mtok * DIn;     // [M, 33]        675,840
    float* DT   = PROJ + (size_t)Mtok * 33;    // [M, 256]     5,242,880
    float* OUTB = DT + (size_t)Mtok * DIn;     // [M, DM]      7,884,800
    float* POOL = OUTB + (size_t)Mtok * DMod;  // [B, DM]          6,160

    const size_t nXDM = (size_t)Mtok * DMod;

    zero_slots_kernel<<<1, 32, 0, stream>>>(SLOT);

    // fq(x), fq(pe_w) scales; patch embed with on-the-fly quant; fq(embed)
    amax_kernel<<<1024, 256, 0, stream>>>(x, (size_t)Bn * Hh * Wi, SLOT + 0);
    amax_kernel<<<4, 256, 0, stream>>>(pe_w, (size_t)EMB * 2 * 16, SLOT + 1);
    patch_embed_kernel<<<cdiv((int)nXDM, 256), 256, 0, stream>>>(
        x, pe_w, pe_b, SLOT + 0, SLOT + 1, X, SLOT + 2);
    quant_kernel<<<1024, 256, 0, stream>>>(X, nXDM, SLOT + 2);

    // x = fq(fq(x) + fq(fq(pos)))  (fq is idempotent in forward)
    amax_kernel<<<256, 256, 0, stream>>>(pos, (size_t)Lseq * DMod, SLOT + 3);
    add_pos_kernel<<<cdiv((int)nXDM, 256), 256, 0, stream>>>(X, pos, SLOT + 3, SLOT + 4);
    quant_kernel<<<1024, 256, 0, stream>>>(X, nXDM, SLOT + 4);

    for (int blk = 0; blk < 2; ++blk) {
        const float* inw = in_w + (size_t)blk * DMod * 512;
        const float* cwp = cw + blk * DIn * 4;
        const float* cbp = cb + blk * DIn;
        const float* xpw = xp_w + (size_t)blk * DIn * 33;
        const float* dtw = dt_w + (size_t)blk * DTR * DIn;
        const float* dtb = dt_b + blk * DIn;
        const float* alg = A_log + blk * DIn * 4;
        const float* dpp = Dp + blk * DIn;
        const float* ow  = out_w + (size_t)blk * DIn * DMod;

        // xz = x @ in_w  (shared by both directions; flip only permutes rows)
        {
            dim3 grid(512 / TN, Mtok / TM);
            gemm_wmma_kernel<0><<<grid, 256, 0, stream>>>(X, DMod, inw, XZ,
                                                          Mtok, DMod, 512, nullptr, Lseq);
        }
        for (int rev = 0; rev < 2; ++rev) {
            conv_silu_kernel<<<cdiv(Mtok * DIn, 256), 256, 0, stream>>>(XZ, cwp, cbp, rev, XC);
            {   // proj = xc @ xp_w  -> [M, 33]
                dim3 grid(cdiv(33, TN), Mtok / TM);
                gemm_wmma_kernel<0><<<grid, 256, 0, stream>>>(XC, DIn, xpw, PROJ,
                                                              Mtok, DIn, 33, nullptr, Lseq);
            }
            {   // dt = softplus(dtr @ dt_w + dt_b)   (A = PROJ[:, :25], lda=33)
                dim3 grid(DIn / TN, Mtok / TM);
                gemm_wmma_kernel<2><<<grid, 256, 0, stream>>>(PROJ, 33, dtw, DT,
                                                              Mtok, DTR, DIn, dtb, Lseq);
            }
            scan_kernel<<<Bn, DIn, 0, stream>>>(XC, DT, PROJ, XZ, alg, dpp, rev);
            // out = y @ out_w ; fwd stores, bwd flip-accumulates (fwd + flip(bwd))
            dim3 grid(cdiv(DMod, TN), Mtok / TM);
            if (rev == 0)
                gemm_wmma_kernel<0><<<grid, 256, 0, stream>>>(XC, DIn, ow, OUTB,
                                                              Mtok, DIn, DMod, nullptr, Lseq);
            else
                gemm_wmma_kernel<3><<<grid, 256, 0, stream>>>(XC, DIn, ow, OUTB,
                                                              Mtok, DIn, DMod, nullptr, Lseq);
        }
        // fq(fwd+bwd), then x = fq(layernorm(x + s))
        int sS = 5 + blk * 2, sL = 6 + blk * 2;
        amax_kernel<<<1024, 256, 0, stream>>>(OUTB, nXDM, SLOT + sS);
        quant_kernel<<<1024, 256, 0, stream>>>(OUTB, nXDM, SLOT + sS);
        ln_kernel<<<Mtok, 128, 0, stream>>>(X, OUTB, ln_g + blk * DMod,
                                            ln_b + blk * DMod, SLOT + sL);
        quant_kernel<<<1024, 256, 0, stream>>>(X, nXDM, SLOT + sL);
    }

    // mean pool, fq, classifier with fq(cls_w)
    pool_kernel<<<cdiv(Bn * DMod, 256), 256, 0, stream>>>(X, POOL);
    amax_kernel<<<4, 256, 0, stream>>>(POOL, (size_t)Bn * DMod, SLOT + 9);
    quant_kernel<<<4, 256, 0, stream>>>(POOL, (size_t)Bn * DMod, SLOT + 9);
    amax_kernel<<<1, 256, 0, stream>>>(cls_w, (size_t)DMod * 2, SLOT + 10);
    cls_kernel<<<1, 32, 0, stream>>>(POOL, cls_w, cls_b, SLOT + 10, out);
}
